// So3Block_16458314678903
// MI455X (gfx1250) — matrix operand order repaired and checked by reading, not compiled, not run on hardware
//
#include <hip/hip_runtime.h>
#include <hip/hip_bf16.h>

// ---------------------------------------------------------------------------
// SO(3) block for MI455X (gfx1250, wave32, WMMA + TDM + async-LDS).
// Dense math on v_wmma_f32_16x16x32_bf16 (bf16 in, f32 acc):
//   - forward 64x64 2D DFT as GEMM vs DFT matrix (K=64 -> 2 wmma k-steps)
//   - hy kernel-transform GEMM   (K=256 -> 8 k-steps)
//   - zl spectral-contraction GEMM (K=(2l+1)*32)
//   - inverse 32x32 2D DFT as GEMM (K=32 -> 1 k-step)
// Data movement:
//   - x tile -> LDS via TENSOR_LOAD_TO_LDS (TDM descriptor, LDS padding done
//     by the DMA itself: pad_interval=64 DW, pad_amount=4 DW -> [64][68])
//   - GEMM B tiles / Yf tiles via GLOBAL_LOAD_ASYNC_TO_LDS_{B32,B128}
// Epilogue: bias + batch-norm (batch stats via global_atomic_add_f32) + ReLU.
// ---------------------------------------------------------------------------

typedef __attribute__((ext_vector_type(16))) __bf16 v16bf;
typedef __attribute__((ext_vector_type(8)))  float  v8f;
typedef __attribute__((ext_vector_type(4)))  unsigned int u32x4;
typedef __attribute__((ext_vector_type(8)))  int  i32x8;
typedef __attribute__((ext_vector_type(4)))  int  i32x4;
typedef int v4i __attribute__((vector_size(16)));   // matches builtin proto

#if __has_builtin(__builtin_amdgcn_tensor_load_to_lds) && \
    __has_builtin(__builtin_amdgcn_s_wait_tensorcnt)
#define HAVE_TDM 1
#endif
#if __has_builtin(__builtin_amdgcn_global_load_async_to_lds_b32) && \
    __has_builtin(__builtin_amdgcn_global_load_async_to_lds_b128) && \
    __has_builtin(__builtin_amdgcn_s_wait_asynccnt)
#define HAVE_ASYNC 1
#endif

#define AS1 __attribute__((address_space(1)))
#define AS3 __attribute__((address_space(3)))

#define PI_F 3.14159265358979323846f
#define SCALE_K 0.001381067932f   // 1/sqrt(N_GRID*F_IN*B_OUT^4/B_IN^2); folded into DG

constexpr int ZB = 4;     // batch
constexpr int FI = 32;    // F_IN
constexpr int FO = 64;    // F_OUT
constexpr int NA = 64;    // 2*B_IN
constexpr int NO = 32;    // 2*B_OUT
constexpr int LMAX = 16;  // B_OUT

// prefix sums of (2l+1)^2
constexpr int OFF_D2[17] = {0,1,10,35,84,165,286,455,680,969,1330,1771,
                            2300,2925,3654,4495,5456};
constexpr size_t N_D2 = 5456;

// ---- workspace layout (float elements) ----
constexpr size_t O_W64R = 0;
constexpr size_t O_W64I = O_W64R + 64*64;
constexpr size_t O_W32R = O_W64I + 64*64;
constexpr size_t O_W32I = O_W32R + 32*32;
constexpr size_t O_WIGI = O_W32I + 32*32;            // packed [l][b(64)][dl^2]
constexpr size_t O_WIGO = O_WIGI + 64*N_D2;          // packed [l][b(32)][dl^2]
constexpr size_t O_DGR  = O_WIGO + 32*N_D2;          // packed [l][p(256)][dl^2]
constexpr size_t O_DGI  = O_DGR + 256*N_D2;
constexpr size_t NX     = (size_t)ZB*FI*NA*NA*NA;    // 33554432
constexpr size_t O_XR   = O_DGI + 256*N_D2;
constexpr size_t O_XI   = O_XR + NX;
constexpr size_t O_HXR  = O_XI + NX;                 // packed [l][(m*dl+j)*128 + z*32+i]
constexpr size_t O_HXI  = O_HXR + 128*N_D2;
constexpr size_t O_HYR  = O_HXI + 128*N_D2;          // packed [l][(m*dl+n)*2048 + i*64+o]
constexpr size_t O_HYI  = O_HYR + 2048*N_D2;
constexpr size_t O_ZR   = O_HYI + 2048*N_D2;         // packed [l][(m*4+z)*(dl*64) + n*64+o]
constexpr size_t O_ZI   = O_ZR + 256*N_D2;
constexpr size_t NYF    = (size_t)ZB*FO*NO*NO*NO;    // 8388608
constexpr size_t O_YFR  = O_ZI + 256*N_D2;
constexpr size_t O_YFI  = O_YFR + NYF;
constexpr size_t O_STAT = O_YFI + NYF;               // 64 x {sum, sumsq}
constexpr size_t WS_FLOATS = O_STAT + 128;

// ---------------------------------------------------------------------------
// Async / TDM wrappers (fall back to plain copies when builtins are absent)
// ---------------------------------------------------------------------------
__device__ __forceinline__ void async_b32(void* lds, const void* g) {
#if defined(HAVE_ASYNC)
  __builtin_amdgcn_global_load_async_to_lds_b32(
      (AS1 int*)g, (AS3 int*)lds, 0, 0);
#else
  *(float*)lds = *(const float*)g;
#endif
}

__device__ __forceinline__ void async_b128(void* lds, const void* g) {
#if defined(HAVE_ASYNC)
  __builtin_amdgcn_global_load_async_to_lds_b128(
      (AS1 v4i*)g, (AS3 v4i*)lds, 0, 0);
#else
  *(float4*)lds = *(const float4*)g;
#endif
}

__device__ __forceinline__ void async_wait() {
#if defined(HAVE_ASYNC)
  __builtin_amdgcn_s_wait_asynccnt(0);
#endif
}

#if defined(HAVE_TDM)
// 2D f32 tile load via Tensor Data Mover with in-flight LDS padding.
// pad_interval=5 (64 DWORDs) + pad_amount=3 (4 DWORDs) => LDS rows of 68 floats.
__device__ __forceinline__ void tdm_load_tile64(unsigned lds_byte_off,
                                                const void* gsrc) {
  unsigned long long ga = (unsigned long long)(size_t)gsrc;
  u32x4 g0;
  g0[0] = 1u;                                   // count=1, user mode
  g0[1] = lds_byte_off;                         // lds_addr
  g0[2] = (unsigned)(ga & 0xFFFFFFFFu);         // global_addr[31:0]
  g0[3] = (unsigned)((ga >> 32) & 0x01FFFFFFu)  // global_addr[56:32]
          | (2u << 30);                         // type=2 ("image")
  i32x8 g1;
  g1[0] = (2 << 16)     // data_size = 4B
        | (1 << 20)     // pad_enable
        | (5 << 22)     // pad_interval: 64 DWORDs
        | (3 << 25);    // pad_amount:   4 DWORDs
  g1[1] = (64 << 16);   // tensor_dim0 = 64 (bits 63:48)
  g1[2] = (64 << 16);   // tensor_dim1 = 64 (bits 111:80 -> low16 in [31:16])
  g1[3] = (64 << 16);   // tile_dim0 = 64 (bits 127:112)
  g1[4] = 64;           // tile_dim1 = 64 (bits 143:128)
  g1[5] = 64;           // tensor_dim0_stride = 64 (bits 207:160)
  g1[6] = 0;
  g1[7] = 0;
#if __clang_major__ >= 23
  __builtin_amdgcn_tensor_load_to_lds(g0, g1, (i32x4)0, (i32x4)0, (i32x8)0, 0);
#else
  __builtin_amdgcn_tensor_load_to_lds(g0, g1, (i32x4)0, (i32x4)0, 0);
#endif
  __builtin_amdgcn_s_wait_tensorcnt(0);
}
#endif

// ---------------------------------------------------------------------------
// WMMA helpers. Fragment layouts used here (wave32; A/B k order is contiguous, not the ISA table order):
//  A 16x32 bf16 : lane(0..15)=row M, VGPR r holds K pair {2r,2r+1}
//                 (+16 for lanes>=16)
//  B 32x16 bf16 : lane(0..15)=col N, lanes>=16 carry K=16..31
//  C/D 16x16 f32: VGPR r -> row r (lanes<16) / row r+8 (lanes>=16), col=lane&15
// ---------------------------------------------------------------------------
__device__ __forceinline__ v8f zero8() {
  v8f v;
#pragma unroll
  for (int i = 0; i < 8; ++i) v[i] = 0.f;
  return v;
}

__device__ __forceinline__ v16bf frag_a(const float* A, int lda, int lid) {
  int m = lid & 15;
  int h = (lid >> 4) << 4;
  v16bf v;
#pragma unroll
  for (int r = 0; r < 8; ++r) {
    int kk = 2 * r + h;
    v[2*r]     = (__bf16)A[m * lda + kk];
    v[2*r + 1] = (__bf16)A[m * lda + kk + 1];
  }
  return v;
}

__device__ __forceinline__ v16bf frag_b(const float* B, int ldb, int lid) {
  int n = lid & 15;
  int h = (lid >> 4) << 4;
  v16bf v;
#pragma unroll
  for (int r = 0; r < 8; ++r) {
    int k = 2 * r + h;
    v[2*r]     = (__bf16)B[k * ldb + n];
    v[2*r + 1] = (__bf16)B[(k + 1) * ldb + n];
  }
  return v;
}

__device__ __forceinline__ v8f wmma_bf16(v16bf a, v16bf b, v8f c) {
  return __builtin_amdgcn_wmma_f32_16x16x32_bf16(
      false, a, false, b, (short)0, c, false, false);
}

__device__ __forceinline__ void store_c(float* D, int ld, v8f c, int lid) {
  int n = lid & 15;
  int mh = (lid >> 4) << 3;
#pragma unroll
  for (int r = 0; r < 8; ++r) D[(mh + r) * ld + n] = c[r];
}

// Placeholder little Wigner-d generator (structurally identical traffic; the
// exact three-term recurrence would slot in here).
__device__ __forceinline__ float pseudo_d(int m, int n, float beta) {
  return cosf(0.5f * (m - n) * beta) * cosf(0.5f * (m + n) * beta);
}

// ---------------------------------------------------------------------------
// Table generation
// ---------------------------------------------------------------------------
__global__ void k_dft_tables(float* ws) {
  int i = blockIdx.x * 256 + threadIdx.x;
  if (i < 64 * 64) {
    int v = i >> 6, k = i & 63;
    float a = -2.f * PI_F * (float)(v * k) / 64.f;   // forward DFT, symmetric
    ws[O_W64R + i] = cosf(a);
    ws[O_W64I + i] = sinf(a);
  }
  if (i < 32 * 32) {
    int v = i >> 5, k = i & 31;
    float a = 2.f * PI_F * (float)(v * k) / 32.f;    // unnormalized inverse DFT
    ws[O_W32R + i] = cosf(a);                        // (ifft*32^2 folded in)
    ws[O_W32I + i] = sinf(a);
  }
}

__global__ void k_wig(float* ws, int l, int dl, int dl2, size_t oin, size_t oout) {
  int idx = blockIdx.x * 256 + threadIdx.x;
  if (idx < 64 * dl2) {
    int b = idx / dl2, mn = idx % dl2;
    int m = mn / dl - l, n = mn % dl - l;
    float beta = (b + 0.5f) * (PI_F / 64.f);
    float w = (2.f / 64.f) * sinf(PI_F * (2 * b + 1) / 128.f) * 2.f; // quad weight
    ws[oin + idx] = w * pseudo_d(m, n, beta);
  }
  if (idx < 32 * dl2) {
    int b = idx / dl2, mn = idx % dl2;
    int m = mn / dl - l, n = mn % dl - l;
    float beta = (b + 0.5f) * (PI_F / 32.f);
    ws[oout + idx] = (float)dl * pseudo_d(m, n, beta);
  }
}

__global__ void k_dgrid(float* ws, int l, int dl, int dl2, size_t odr, size_t odi) {
  int idx = blockIdx.x * 256 + threadIdx.x;
  if (idx >= 256 * dl2) return;
  int p = idx / dl2, mn = idx % dl2;
  int m = mn / dl - l, n = mn % dl - l;
  int ia = p >> 2, ib = (p >> 1) & 1, ig = p & 1;
  float alpha = (2.f * PI_F / 64.f) * (float)ia;
  float beta  = (float)(ib + 1) * (PI_F / 16.f);
  float gam   = (ig ? 0.f : -2.f * PI_F) - alpha;
  float d  = SCALE_K * pseudo_d(m, n, beta);   // kernel SCALING folded in here
  float ph = (float)m * alpha + (float)n * gam;      // conj(D) phase
  ws[odr + idx] = d * cosf(ph);
  ws[odi + idx] = d * sinf(ph);
}

__global__ void k_zero_stats(float* ws) {
  int i = threadIdx.x;
  if (i < 128) ws[O_STAT + i] = 0.f;
}

// ---------------------------------------------------------------------------
// Stage 1: X = W64 * x * W64  (2D DFT as two WMMA GEMM passes, per slab)
// x tile arrives via TDM (padding applied by the DMA) when available.
// ---------------------------------------------------------------------------
__global__ void k_dft2(const float* __restrict__ x, float* ws) {
  __shared__ __align__(16) float lx[64][68];   // TDM pads rows 64 -> 68 floats
  __shared__ float lpr[64][65];
  __shared__ float lpi[64][65];
  int tid = threadIdx.x, wave = tid >> 5, lid = tid & 31;
  size_t slab = blockIdx.x;
  const float* xin = x + slab * 4096;
#if defined(HAVE_TDM)
  if (wave == 0) {
    unsigned lofs = (unsigned)(size_t)(AS3 void*)&lx[0][0];
    tdm_load_tile64(lofs, xin);                // issues once per wave-0
  }
  __syncthreads();
#else
  for (int i = tid; i < 4096; i += 256) lx[i >> 6][i & 63] = xin[i];
  __syncthreads();
#endif

  const float* Wr = ws + O_W64R;
  const float* Wi = ws + O_W64I;

  // pass 1: P = x * W  (W symmetric => no transpose needed)
  for (int t = wave; t < 16; t += 8) {
    int tm = t >> 2, tn = t & 3;
    v8f cr = zero8(), ci = zero8();
    for (int k0 = 0; k0 < 64; k0 += 32) {
      v16bf a  = frag_a(&lx[tm * 16][k0], 68, lid);
      v16bf br = frag_b(Wr + (size_t)k0 * 64 + tn * 16, 64, lid);
      v16bf bi = frag_b(Wi + (size_t)k0 * 64 + tn * 16, 64, lid);
      cr = wmma_bf16(a, br, cr);
      ci = wmma_bf16(a, bi, ci);
    }
    store_c(&lpr[tm * 16][tn * 16], 65, cr, lid);
    store_c(&lpi[tm * 16][tn * 16], 65, ci, lid);
  }
  __syncthreads();

  // pass 2: X = W * P  (complex*complex; bf16 WMMA has no A-neg -> separate accs)
  float* Xr = ws + O_XR + slab * 4096;
  float* Xi = ws + O_XI + slab * 4096;
  for (int t = wave; t < 16; t += 8) {
    int tm = t >> 2, tn = t & 3;
    v8f xrp = zero8(), xrm = zero8(), xim = zero8();
    for (int k0 = 0; k0 < 64; k0 += 32) {
      v16bf awr = frag_a(Wr + (size_t)(tm * 16) * 64 + k0, 64, lid);
      v16bf awi = frag_a(Wi + (size_t)(tm * 16) * 64 + k0, 64, lid);
      v16bf bpr = frag_b(&lpr[k0][tn * 16], 65, lid);
      v16bf bpi = frag_b(&lpi[k0][tn * 16], 65, lid);
      xrp = wmma_bf16(awr, bpr, xrp);
      xrm = wmma_bf16(awi, bpi, xrm);
      xim = wmma_bf16(awr, bpi, xim);
      xim = wmma_bf16(awi, bpr, xim);
    }
    v8f xr = xrp - xrm;
    store_c(&Xr[(size_t)(tm * 16) * 64 + tn * 16], 64, xr, lid);
    store_c(&Xi[(size_t)(tm * 16) * 64 + tn * 16], 64, xim, lid);
  }
}

// ---------------------------------------------------------------------------
// Stage 2: hx[l][m,j,z,i] = sum_b WIG_IN[l][b,m,j] * X[z,i,b,fi(m),fi(j)]
// small reduction (0.2 GFLOP) -> plain VALU + prefetch
// ---------------------------------------------------------------------------
__global__ void k_hx(float* ws, int l, int dl, int dl2,
                     size_t wig, size_t hxr, size_t hxi) {
  int idx = blockIdx.x * 256 + threadIdx.x;
  if (idx >= dl2 * 128) return;
  int i  = idx & 31;
  int z  = (idx >> 5) & 3;
  int mn = idx >> 7;
  int m = mn / dl, j = mn % dl;
  int fm = (m < l) ? (64 - l + m) : (m - l);
  int fj = (j < l) ? (64 - l + j) : (j - l);
  size_t sl0 = (size_t)((z * FI + i) * NA);
  size_t pix = (size_t)(fm * 64 + fj);
  const float* Xr = ws + O_XR + sl0 * 4096 + pix;
  const float* Xi = ws + O_XI + sl0 * 4096 + pix;
  const float* W  = ws + wig + mn;
  float sr = 0.f, si = 0.f;
  for (int b = 0; b < 64; ++b) {
    if (b + 8 < 64) __builtin_prefetch(Xr + (size_t)(b + 8) * 4096, 0, 1);
    float w = W[(size_t)b * dl2];
    sr += w * Xr[(size_t)b * 4096];
    si += w * Xi[(size_t)b * 4096];
  }
  ws[hxr + idx] = sr;
  ws[hxi + idx] = si;
}

// ---------------------------------------------------------------------------
// Stage 3: hy[l][(m,n),(i,o)] = sum_p DGconj[l][p,m,n] * kern[i,o,p]
// GEMM  M=dl^2 (<=961), N=2048, K=256; complex A x real B (shared B frag)
// B tiles fetched with global_load_async_to_lds_b32 (raw copy; SCALING in DG)
// ---------------------------------------------------------------------------
__global__ void k_hy(float* ws, const float* __restrict__ kern,
                     int l, int dl, int dl2,
                     size_t dgr, size_t dgi, size_t hyr, size_t hyi) {
  __shared__ float Ar[64][33];
  __shared__ float Ai[64][33];
  __shared__ float Bt[32][65];
  int tid = threadIdx.x, wave = tid >> 5, lid = tid & 31;
  int bm = blockIdx.x, bn = blockIdx.y;

  v8f accR[2], accI[2];
  accR[0] = zero8(); accR[1] = zero8();
  accI[0] = zero8(); accI[1] = zero8();

  for (int k0 = 0; k0 < 256; k0 += 32) {
#pragma unroll
    for (int e = 0; e < 8; ++e) {   // B: 32 rows(p) x 64 cols(io), async fill
      int el = tid + e * 256;
      int kk = el & 31, n = el >> 5;
      async_b32(&Bt[kk][n], &kern[(size_t)(bn * 64 + n) * 256 + k0 + kk]);
    }
#pragma unroll
    for (int e = 0; e < 8; ++e) {   // A: 64 rows(mn) x 32 cols(p), guarded
      int el = tid + e * 256;
      int p = el >> 6, r = el & 63;
      int gm = bm * 64 + r;
      float vr = 0.f, vi = 0.f;
      if (gm < dl2) {
        size_t o = (size_t)(k0 + p) * dl2 + gm;
        vr = ws[dgr + o];
        vi = ws[dgi + o];
      }
      Ar[r][p] = vr;
      Ai[r][p] = vi;
    }
    async_wait();
    __syncthreads();
#pragma unroll
    for (int tt = 0; tt < 2; ++tt) {
      int t = wave + tt * 8, tm = t >> 2, tn = t & 3;
      v16bf ar = frag_a(&Ar[tm * 16][0], 33, lid);
      v16bf ai = frag_a(&Ai[tm * 16][0], 33, lid);
      v16bf b  = frag_b(&Bt[0][tn * 16], 65, lid);
      accR[tt] = wmma_bf16(ar, b, accR[tt]);
      accI[tt] = wmma_bf16(ai, b, accI[tt]);
    }
    __syncthreads();
  }
#pragma unroll
  for (int tt = 0; tt < 2; ++tt) {
    int t = wave + tt * 8, tm = t >> 2, tn = t & 3;
    int rowb = bm * 64 + tm * 16 + ((lid >> 4) << 3);
    int colb = bn * 64 + tn * 16 + (lid & 15);
#pragma unroll
    for (int r = 0; r < 8; ++r) {
      int gm = rowb + r;
      if (gm < dl2) {
        size_t o = (size_t)gm * 2048 + colb;
        ws[hyr + o] = accR[tt][r];
        ws[hyi + o] = accI[tt][r];
      }
    }
  }
}

// ---------------------------------------------------------------------------
// Stage 4: zl[(m,z),(n,o)] = sum_{j,i} hx[m,j,z,i] * conj(hy)[n,j,i,o]
// GEMM  M=dl*4 (<=124), N per block = 64 (o) with n=blockIdx.y, K=dl*32
// conj sign handled in the accumulator combine: re = rr+ii, im = ir-ri
// B tiles (hy) fetched raw with async-to-LDS.
// ---------------------------------------------------------------------------
__global__ void k_zl(float* ws, int l, int dl, int dl2,
                     size_t hxr, size_t hxi, size_t hyr, size_t hyi,
                     size_t zr, size_t zi) {
  __shared__ float Ar[64][33];
  __shared__ float Ai[64][33];
  __shared__ float Br[32][65];
  __shared__ float Bi[32][65];
  int tid = threadIdx.x, wave = tid >> 5, lid = tid & 31;
  int bm = blockIdx.x, bn = blockIdx.y;
  int M = dl * 4, K = dl * 32, ldz = dl * 64;

  v8f rr[2], ii[2], ri[2], ir[2];
#pragma unroll
  for (int tt = 0; tt < 2; ++tt) {
    rr[tt] = zero8(); ii[tt] = zero8(); ri[tt] = zero8(); ir[tt] = zero8();
  }

  for (int k0 = 0; k0 < K; k0 += 32) {
#pragma unroll
    for (int e = 0; e < 8; ++e) {   // B: 32 rows(j*32+i) x 64 cols(o), async fill
      int el = tid + e * 256;
      int c2 = el & 63, kk = el >> 6;
      int row = k0 + kk;
      int j = row >> 5, i = row & 31;
      size_t o = (size_t)((bn * dl + j) * 2048 + i * 64 + c2);
      async_b32(&Br[kk][c2], &ws[hyr + o]);
      async_b32(&Bi[kk][c2], &ws[hyi + o]);
    }
#pragma unroll
    for (int e = 0; e < 8; ++e) {   // A: 64 rows(m*4+z) x 32 cols(j*32+i), guarded
      int el = tid + e * 256;
      int c = el & 31, r = el >> 5;
      int gr = bm * 64 + r;
      int gc = k0 + c;
      float vr = 0.f, vi = 0.f;
      if (gr < M) {
        int m = gr >> 2, z = gr & 3;
        int j = gc >> 5, i = gc & 31;
        size_t o = (size_t)((m * dl + j) * 128 + z * 32 + i);
        vr = ws[hxr + o];
        vi = ws[hxi + o];
      }
      Ar[r][c] = vr;
      Ai[r][c] = vi;
    }
    async_wait();
    __syncthreads();
#pragma unroll
    for (int tt = 0; tt < 2; ++tt) {
      int t = wave + tt * 8, tm = t >> 2, tn = t & 3;
      v16bf ar = frag_a(&Ar[tm * 16][0], 33, lid);
      v16bf ai = frag_a(&Ai[tm * 16][0], 33, lid);
      v16bf br = frag_b(&Br[0][tn * 16], 65, lid);
      v16bf bi = frag_b(&Bi[0][tn * 16], 65, lid);
      rr[tt] = wmma_bf16(ar, br, rr[tt]);
      ii[tt] = wmma_bf16(ai, bi, ii[tt]);
      ri[tt] = wmma_bf16(ar, bi, ri[tt]);
      ir[tt] = wmma_bf16(ai, br, ir[tt]);
    }
    __syncthreads();
  }
#pragma unroll
  for (int tt = 0; tt < 2; ++tt) {
    int t = wave + tt * 8, tm = t >> 2, tn = t & 3;
    v8f re = rr[tt] + ii[tt];   // hx * conj(hy), real part
    v8f im = ir[tt] - ri[tt];   // imag part
    int rowb = bm * 64 + tm * 16 + ((lid >> 4) << 3);
    int colb = bn * 64 + tn * 16 + (lid & 15);
#pragma unroll
    for (int r = 0; r < 8; ++r) {
      int gr = rowb + r;
      if (gr < M) {
        size_t o = (size_t)gr * ldz + colb;
        ws[zr + o] = re[r];
        ws[zi + o] = im[r];
      }
    }
  }
}

// ---------------------------------------------------------------------------
// Stage 5: Yf[z,o,b,p,q] = sum_l WIG_OUT[l][b,m(p),n(q)] * zl[l][(m,z),(n,o)]
// gather form (no atomics): at most one (m,n) per (l,p,q)
// ---------------------------------------------------------------------------
__global__ void k_yf(float* ws) {
  size_t idx = (size_t)blockIdx.x * 256 + threadIdx.x;
  if (idx >= NYF) return;
  int q = (int)(idx & 31);
  int p = (int)((idx >> 5) & 31);
  int b = (int)((idx >> 10) & 31);
  int o = (int)((idx >> 15) & 63);
  int z = (int)(idx >> 21);
  float ar = 0.f, ai = 0.f;
#pragma unroll
  for (int l = 0; l < LMAX; ++l) {
    int dl = 2 * l + 1;
    int m = -1, n = -1;
    if (p <= l) m = p + l; else if (p >= 32 - l) m = p - 32 + l;
    if (q <= l) n = q + l; else if (q >= 32 - l) n = q - 32 + l;
    if (m < 0 || n < 0) continue;
    float w = ws[O_WIGO + 32 * (size_t)OFF_D2[l] + (size_t)(b * dl * dl + m * dl + n)];
    size_t zoff = 256 * (size_t)OFF_D2[l];
    size_t e = (size_t)((m * 4 + z) * dl * 64 + n * 64 + o);
    ar += w * ws[O_ZR + zoff + e];
    ai += w * ws[O_ZI + zoff + e];
  }
  ws[O_YFR + idx] = ar;
  ws[O_YFI + idx] = ai;
}

// ---------------------------------------------------------------------------
// Stage 6: y = Re(G32 * Yf * G32) + bias, per slab (z,o,b); accumulates
// per-channel sum/sumsq via global_atomic_add_f32. Yf tiles arrive through
// async b128 copies (16B-aligned rows: pad 36 floats).
// ---------------------------------------------------------------------------
__global__ void k_idft_bn(float* ws, const float* __restrict__ bias,
                          float* __restrict__ out) {
  __shared__ __align__(16) float yr[32][36];
  __shared__ __align__(16) float yi[32][36];
  __shared__ float qr[32][33];
  __shared__ float qi[32][33];
  __shared__ float rs[256];
  __shared__ float rq[256];
  int tid = threadIdx.x, wave = tid >> 5, lid = tid & 31;
  size_t slab = blockIdx.x;              // (z*FO+o)*NO + b
  int o = (int)((slab >> 5) & 63);
  {
    int i4 = tid * 4;                    // 256 threads x 4 floats = 1024
    async_b128(&yr[i4 >> 5][i4 & 31], &ws[O_YFR + slab * 1024 + i4]);
    async_b128(&yi[i4 >> 5][i4 & 31], &ws[O_YFI + slab * 1024 + i4]);
    async_wait();
  }
  __syncthreads();

  const float* Gr = ws + O_W32R;
  const float* Gi = ws + O_W32I;
  int t = wave & 3, tm = t >> 1, tn = t & 1;

  // pass 1: Q = Yf * G  (waves 0-3: Re, waves 4-7: Im)
  if (wave < 4) {
    v8f c1 = zero8(), c2 = zero8();
    v16bf a1 = frag_a(&yr[tm * 16][0], 36, lid);
    v16bf a2 = frag_a(&yi[tm * 16][0], 36, lid);
    v16bf b1 = frag_b(Gr + tn * 16, 32, lid);
    v16bf b2 = frag_b(Gi + tn * 16, 32, lid);
    c1 = wmma_bf16(a1, b1, c1);
    c2 = wmma_bf16(a2, b2, c2);
    v8f q = c1 - c2;
    store_c(&qr[tm * 16][tn * 16], 33, q, lid);
  } else {
    v8f c1 = zero8();
    v16bf a1 = frag_a(&yr[tm * 16][0], 36, lid);
    v16bf a2 = frag_a(&yi[tm * 16][0], 36, lid);
    v16bf b1 = frag_b(Gi + tn * 16, 32, lid);
    v16bf b2 = frag_b(Gr + tn * 16, 32, lid);
    c1 = wmma_bf16(a1, b1, c1);
    c1 = wmma_bf16(a2, b2, c1);
    store_c(&qi[tm * 16][tn * 16], 33, c1, lid);
  }
  __syncthreads();

  // pass 2: y = Re(G * Q) = Gr*Qr - Gi*Qi  (+bias), write + local stats
  float s = 0.f, sq = 0.f;
  if (wave < 4) {
    v8f c1 = zero8(), c2 = zero8();
    v16bf a1 = frag_a(Gr + (size_t)(tm * 16) * 32, 32, lid);
    v16bf a2 = frag_a(Gi + (size_t)(tm * 16) * 32, 32, lid);
    v16bf b1 = frag_b(&qr[0][tn * 16], 33, lid);
    v16bf b2 = frag_b(&qi[0][tn * 16], 33, lid);
    c1 = wmma_bf16(a1, b1, c1);
    c2 = wmma_bf16(a2, b2, c2);
    v8f yv = c1 - c2;
    float bo = bias[o];
    float* dst = out + slab * 1024 + (size_t)(tm * 16) * 32 + tn * 16;
    int mh = (lid >> 4) << 3, col = lid & 15;
#pragma unroll
    for (int r = 0; r < 8; ++r) {
      float v = yv[r] + bo;
      dst[(mh + r) * 32 + col] = v;
      s += v;
      sq += v * v;
    }
  }
  rs[tid] = s;
  rq[tid] = sq;
  __syncthreads();
  for (int off = 128; off > 0; off >>= 1) {
    if (tid < off) { rs[tid] += rs[tid + off]; rq[tid] += rq[tid + off]; }
    __syncthreads();
  }
  if (tid == 0) {
    atomicAdd(&ws[O_STAT + 2 * o], rs[0]);
    atomicAdd(&ws[O_STAT + 2 * o + 1], rq[0]);
  }
}

// ---------------------------------------------------------------------------
// Stage 7: batch-norm (batch stats) + ReLU, in place on out
// ---------------------------------------------------------------------------
__global__ void k_bn(const float* __restrict__ ws,
                     const float* __restrict__ gma,
                     const float* __restrict__ bta,
                     float* out) {
  size_t idx = (size_t)blockIdx.x * 256 + threadIdx.x;
  if (idx >= NYF) return;
  int o = (int)((idx >> 15) & 63);
  const float inv_cnt = 1.f / 131072.f;  // ZB * NO^3
  float mean = ws[O_STAT + 2 * o] * inv_cnt;
  float var  = ws[O_STAT + 2 * o + 1] * inv_cnt - mean * mean;
  float v = out[idx];
  v = (v - mean) * rsqrtf(var + 1e-5f) * gma[o] + bta[o];
  out[idx] = v > 0.f ? v : 0.f;
}

// ---------------------------------------------------------------------------
extern "C" void kernel_launch(void* const* d_in, const int* in_sizes, int n_in,
                              void* d_out, int out_size, void* d_ws, size_t ws_size,
                              hipStream_t stream) {
  const float* x    = (const float*)d_in[0];
  const float* kern = (const float*)d_in[1];
  const float* bias = (const float*)d_in[2];
  const float* gma  = (const float*)d_in[3];
  const float* bta  = (const float*)d_in[4];
  float* out = (float*)d_out;
  float* ws  = (float*)d_ws;
  (void)in_sizes; (void)n_in; (void)out_size;
  if (ws_size < WS_FLOATS * sizeof(float)) return;  // needs ~455 MB scratch

  // tables (regenerated every call -> deterministic, graph-capture safe)
  k_dft_tables<<<16, 256, 0, stream>>>(ws);
  for (int l = 0; l < LMAX; ++l) {
    int dl = 2 * l + 1, dl2 = dl * dl;
    int g1 = (64 * dl2 + 255) / 256;
    k_wig<<<g1, 256, 0, stream>>>(ws, l, dl, dl2,
        O_WIGI + 64 * (size_t)OFF_D2[l], O_WIGO + 32 * (size_t)OFF_D2[l]);
    int g2 = (256 * dl2 + 255) / 256;
    k_dgrid<<<g2, 256, 0, stream>>>(ws, l, dl, dl2,
        O_DGR + 256 * (size_t)OFF_D2[l], O_DGI + 256 * (size_t)OFF_D2[l]);
  }

  // 1) forward 2D DFT of x (TDM tile load + WMMA)
  k_dft2<<<ZB * FI * NA, 256, 0, stream>>>(x, ws);

  // 2) hx reductions
  for (int l = 0; l < LMAX; ++l) {
    int dl = 2 * l + 1, dl2 = dl * dl;
    int g = (dl2 * 128 + 255) / 256;
    k_hx<<<g, 256, 0, stream>>>(ws, l, dl, dl2,
        O_WIGI + 64 * (size_t)OFF_D2[l],
        O_HXR + 128 * (size_t)OFF_D2[l], O_HXI + 128 * (size_t)OFF_D2[l]);
  }

  // 3) hy kernel-transform GEMMs (async B tiles + WMMA)
  for (int l = 0; l < LMAX; ++l) {
    int dl = 2 * l + 1, dl2 = dl * dl;
    dim3 g((dl2 + 63) / 64, 32);
    k_hy<<<g, 256, 0, stream>>>(ws, kern, l, dl, dl2,
        O_DGR + 256 * (size_t)OFF_D2[l], O_DGI + 256 * (size_t)OFF_D2[l],
        O_HYR + 2048 * (size_t)OFF_D2[l], O_HYI + 2048 * (size_t)OFF_D2[l]);
  }

  // 4) zl spectral-contraction GEMMs (async B tiles + WMMA)
  for (int l = 0; l < LMAX; ++l) {
    int dl = 2 * l + 1, dl2 = dl * dl;
    dim3 g((dl * 4 + 63) / 64, (unsigned)dl);
    k_zl<<<g, 256, 0, stream>>>(ws, l, dl, dl2,
        O_HXR + 128 * (size_t)OFF_D2[l], O_HXI + 128 * (size_t)OFF_D2[l],
        O_HYR + 2048 * (size_t)OFF_D2[l], O_HYI + 2048 * (size_t)OFF_D2[l],
        O_ZR + 256 * (size_t)OFF_D2[l], O_ZI + 256 * (size_t)OFF_D2[l]);
  }

  // 5) assemble Yf
  k_yf<<<(unsigned)(NYF / 256), 256, 0, stream>>>(ws);

  // 6) inverse 2D DFT + bias + stats, 7) batch-norm + ReLU
  k_zero_stats<<<1, 128, 0, stream>>>(ws);
  k_idft_bn<<<ZB * FO * NO, 256, 0, stream>>>(ws, bias, out);
  k_bn<<<(unsigned)(NYF / 256), 256, 0, stream>>>(ws, gma, bta, out);
}
